// GraphEncoderOneHead_14834817040600
// MI455X (gfx1250) — matrix operand order repaired and checked by reading, not compiled, not run on hardware
//
#include <hip/hip_runtime.h>
#include <math.h>

typedef __attribute__((ext_vector_type(16))) _Float16 v16h;
typedef __attribute__((ext_vector_type(2)))  _Float16 h2;
typedef __attribute__((ext_vector_type(8)))  float    v8f;

#define BN_EPS 1e-5f

// ============================ helpers ============================

__device__ __forceinline__ float lrelu02(float x) { return x > 0.f ? x : 0.2f * x; }

// float atomic max via signed-max (positives) / unsigned-min (negatives) lattice
__device__ __forceinline__ void atomicMaxF(float* addr, float val) {
  if (val >= 0.f) atomicMax((int*)addr, __float_as_int(val));
  else            atomicMin((unsigned int*)addr, __float_as_uint(val));
}

__global__ void k_fill(float* p, float v, int n) {
  int i = blockIdx.x * blockDim.x + threadIdx.x;
  if (i < n) p[i] = v;
}

// ============================ degree / norm ============================

__global__ void k_deg_count(const int* __restrict__ dst, float* deg, int E) {
  int e = blockIdx.x * blockDim.x + threadIdx.x;
  if (e < E) atomicAdd(&deg[dst[e]], 1.0f);
}

__global__ void k_rsqrt_inplace(float* p, int n) {
  int i = blockIdx.x * blockDim.x + threadIdx.x;
  if (i < n) p[i] = rsqrtf(fmaxf(p[i], 1.0f));
}

// ============================ WMMA GEMM ============================
// C[M,N] = act(A[M,K] @ W[K,N] + bias), fp32 in memory, f16 WMMA math.
// block = 256 threads (8 waves). Block tile: 16 rows x 128 cols; wave w owns cols [16w,16w+16).
// grid = (M/16, N/128). Requires M%16==0, N%128==0, K%32==0.
__global__ __launch_bounds__(256) void k_gemm_wmma(
    const float* __restrict__ A, const float* __restrict__ W,
    const float* __restrict__ bias, float* __restrict__ C,
    int M, int N, int K, int relu)
{
  __shared__ _Float16 sA[16 * 32];    // [row][k]
  __shared__ _Float16 sB[128 * 32];   // transposed: [n][k]

  const int tid  = threadIdx.x;
  const int wave = tid >> 5;
  const int lane = tid & 31;
  const int rowBase = blockIdx.x * 16;
  const int colBase = blockIdx.y * 128;

  const int g = lane >> 4;     // half-wave
  const int m = lane & 15;     // row (A) / col (B,C) within tile

  // staging coordinates (fixed per thread)
  const int ar = tid >> 4;            // A row 0..15
  const int ak = (tid & 15) * 2;      // A k pair 0..30
  const int bn = (tid * 4) & 127;     // B col group start

  v8f acc = {};

  for (int k0 = 0; k0 < K; k0 += 32) {
    __syncthreads();
    // ---- stage A chunk: thread t loads float2 (row ar, k ak..ak+1), packs half2 ----
    {
      const float2 af = *(const float2*)&A[(size_t)(rowBase + ar) * K + (k0 + ak)];
      *(h2*)&sA[ar * 32 + ak] = (h2){(_Float16)af.x, (_Float16)af.y};
    }
    // ---- stage W chunk transposed: 4 x float4 coalesced loads, b16 transposed stores ----
    #pragma unroll
    for (int j = 0; j < 4; ++j) {
      const int f  = j * 1024 + tid * 4;  // flat index into 32x128 chunk
      const int kk = f >> 7;              // 0..31
      const float4 wf = *(const float4*)&W[(size_t)(k0 + kk) * N + (colBase + bn)];
      sB[(bn + 0) * 32 + kk] = (_Float16)wf.x;
      sB[(bn + 1) * 32 + kk] = (_Float16)wf.y;
      sB[(bn + 2) * 32 + kk] = (_Float16)wf.z;
      sB[(bn + 3) * 32 + kk] = (_Float16)wf.w;
    }
    __syncthreads();

    // A fragment: VGPR v holds K pair; k = (v>=4?16:0) + g*8 + 2*(v&3)
    v16h a, b;
    #pragma unroll
    for (int v = 0; v < 8; ++v) {
      int ka = ((v & 4) ? 16 : 0) + g * 8 + 2 * (v & 3);
      a[2 * v]     = sA[m * 32 + ka];
      a[2 * v + 1] = sA[m * 32 + ka + 1];
    }
    // B fragment: same K order as A; k = (v>=4?16:0) + g*8 + 2*(v&3)
    #pragma unroll
    for (int v = 0; v < 8; ++v) {
      int kb = ((v & 4) ? 16 : 0) + g * 8 + 2 * (v & 3);
      b[2 * v]     = sB[(wave * 16 + m) * 32 + kb];
      b[2 * v + 1] = sB[(wave * 16 + m) * 32 + kb + 1];
    }
    acc = __builtin_amdgcn_wmma_f32_16x16x32_f16(
        false, a, false, b, (short)0, acc, false, false);
  }

  // C/D layout: VGPR v -> row (v + 8*g), col = lane&15 within wave tile
  const int c = colBase + wave * 16 + m;
  const float bb = bias[c];
  #pragma unroll
  for (int v = 0; v < 8; ++v) {
    int r = rowBase + v + 8 * g;
    float val = acc[v] + bb;
    if (relu) val = fmaxf(val, 0.f);
    C[(size_t)r * N + c] = val;
  }
}

// ============================ GCN message passing ============================

// Y[i][c] = b[c] + H[i][c] * dinv[i]^2   (self-loop + bias + zero-init)
__global__ void k_gcn_init(const float* __restrict__ H, const float* __restrict__ dinv,
                           const float* __restrict__ b, float* __restrict__ Y, int Nn, int C)
{
  int i = blockIdx.x * blockDim.x + threadIdx.x;
  if (i < Nn * C) {
    int r = i / C, c = i - r * C;
    float di = dinv[r];
    Y[i] = b[c] + H[i] * di * di;
  }
}

// one wave per edge: Y[dst] += H[src] * dinv[src]*dinv[dst]
__global__ __launch_bounds__(256) void k_gcn_scatter(
    const float* __restrict__ H, const int* __restrict__ src, const int* __restrict__ dst,
    const float* __restrict__ dinv, float* __restrict__ Y, int E, int C)
{
  int e = blockIdx.x * 8 + (threadIdx.x >> 5);
  int lane = threadIdx.x & 31;
  if (e >= E) return;
  int s = src[e], d = dst[e];
  if (e + 8 < E) {
    // prefetch next edge's source row (global_prefetch_b8)
    __builtin_prefetch(&H[(size_t)src[e + 8] * C], 0, 0);
  }
  float nrm = dinv[s] * dinv[d];
  float4 hv = ((const float4*)(H + (size_t)s * C))[lane];
  float* yp = Y + (size_t)d * C + lane * 4;
  atomicAdd(yp + 0, hv.x * nrm);
  atomicAdd(yp + 1, hv.y * nrm);
  atomicAdd(yp + 2, hv.z * nrm);
  atomicAdd(yp + 3, hv.w * nrm);
}

// ============================ BatchNorm (training stats) ============================

// blockDim.x == C; each thread owns one channel, sums over a row slab
__global__ void k_bn_stats(const float* __restrict__ Y, float* sums, float* sumsq,
                           int Nn, int C, int rowsPerBlock)
{
  int c = threadIdx.x;
  int r0 = blockIdx.x * rowsPerBlock;
  int r1 = min(r0 + rowsPerBlock, Nn);
  float s = 0.f, q = 0.f;
  for (int r = r0; r < r1; ++r) {
    float v = Y[(size_t)r * C + c];
    s += v; q += v * v;
  }
  atomicAdd(&sums[c], s);
  atomicAdd(&sumsq[c], q);
}

// Xcur[i] = relu((Y[i]-m)*rsqrt(v+eps)*g + be) + Xcur[i]
__global__ void k_bn_apply(const float* __restrict__ Y, const float* __restrict__ sums,
                           const float* __restrict__ sumsq, const float* __restrict__ gam,
                           const float* __restrict__ bet, float* __restrict__ Xcur,
                           int Nn, int C)
{
  int i = blockIdx.x * blockDim.x + threadIdx.x;
  if (i < Nn * C) {
    int c = i % C;
    float inv_n = 1.0f / (float)Nn;
    float mean = sums[c] * inv_n;
    float var = sumsq[c] * inv_n - mean * mean;
    float val = (Y[i] - mean) * rsqrtf(var + BN_EPS) * gam[c] + bet[c];
    Xcur[i] = fmaxf(val, 0.f) + Xcur[i];
  }
}

// ============================ GAT ============================

// per-node attention logits: es=dot(H[i],asrc), ed=dot(H[i],adst); wave per node (C==128)
__global__ __launch_bounds__(256) void k_att(
    const float* __restrict__ H, const float* __restrict__ asrc, const float* __restrict__ adst,
    float* __restrict__ es, float* __restrict__ ed, int Nn, int C)
{
  int node = blockIdx.x * 8 + (threadIdx.x >> 5);
  int lane = threadIdx.x & 31;
  if (node >= Nn) return;
  float4 hv = ((const float4*)(H + (size_t)node * C))[lane];
  float4 a1 = ((const float4*)asrc)[lane];
  float4 a2 = ((const float4*)adst)[lane];
  float s1 = hv.x * a1.x + hv.y * a1.y + hv.z * a1.z + hv.w * a1.w;
  float s2 = hv.x * a2.x + hv.y * a2.y + hv.z * a2.z + hv.w * a2.w;
  #pragma unroll
  for (int off = 16; off > 0; off >>= 1) {
    s1 += __shfl_down(s1, off, 32);
    s2 += __shfl_down(s2, off, 32);
  }
  if (lane == 0) { es[node] = s1; ed[node] = s2; }
}

__global__ void k_edge_max(const float* __restrict__ es, const float* __restrict__ ed,
                           const int* __restrict__ src, const int* __restrict__ dst,
                           float* maxb, int E, int Nn)
{
  int e = blockIdx.x * blockDim.x + threadIdx.x;
  if (e >= E + Nn) return;
  int s = (e < E) ? src[e] : (e - E);
  int d = (e < E) ? dst[e] : (e - E);
  atomicMaxF(&maxb[d], lrelu02(es[s] + ed[d]));
}

__global__ void k_edge_expsum(const float* __restrict__ es, const float* __restrict__ ed,
                              const int* __restrict__ src, const int* __restrict__ dst,
                              const float* __restrict__ maxb, float* sumb, int E, int Nn)
{
  int e = blockIdx.x * blockDim.x + threadIdx.x;
  if (e >= E + Nn) return;
  int s = (e < E) ? src[e] : (e - E);
  int d = (e < E) ? dst[e] : (e - E);
  atomicAdd(&sumb[d], expf(lrelu02(es[s] + ed[d]) - maxb[d]));
}

// one wave per edge (incl self-loops): Y[d] += H[s] * alpha
__global__ __launch_bounds__(256) void k_gat_scatter(
    const float* __restrict__ H, const float* __restrict__ es, const float* __restrict__ ed,
    const int* __restrict__ src, const int* __restrict__ dst,
    const float* __restrict__ maxb, const float* __restrict__ sumb,
    float* __restrict__ Y, int E, int Nn, int C)
{
  int e = blockIdx.x * 8 + (threadIdx.x >> 5);
  int lane = threadIdx.x & 31;
  if (e >= E + Nn) return;
  int s = (e < E) ? src[e] : (e - E);
  int d = (e < E) ? dst[e] : (e - E);
  float alpha = expf(lrelu02(es[s] + ed[d]) - maxb[d]) / (sumb[d] + 1e-16f);
  float4 hv = ((const float4*)(H + (size_t)s * C))[lane];
  float* yp = Y + (size_t)d * C + lane * 4;
  atomicAdd(yp + 0, hv.x * alpha);
  atomicAdd(yp + 1, hv.y * alpha);
  atomicAdd(yp + 2, hv.z * alpha);
  atomicAdd(yp + 3, hv.w * alpha);
}

__global__ void k_broadcast_bias(const float* __restrict__ b, float* __restrict__ Y, int Nn, int C)
{
  int i = blockIdx.x * blockDim.x + threadIdx.x;
  if (i < Nn * C) Y[i] = b[i % C];
}

// ============================ pooling ============================

__global__ __launch_bounds__(256) void k_pool_scatter(
    const float* __restrict__ X, const int* __restrict__ batch,
    float* __restrict__ pool, float* __restrict__ cnt, int Nn, int C)
{
  int node = blockIdx.x * 8 + (threadIdx.x >> 5);
  int lane = threadIdx.x & 31;
  if (node >= Nn) return;
  int b = batch[node];
  float4 v = ((const float4*)(X + (size_t)node * C))[lane];
  float* pp = pool + (size_t)b * C + lane * 4;
  atomicAdd(pp + 0, v.x);
  atomicAdd(pp + 1, v.y);
  atomicAdd(pp + 2, v.z);
  atomicAdd(pp + 3, v.w);
  if (lane == 0) atomicAdd(&cnt[b], 1.0f);
}

__global__ void k_pool_div(const float* __restrict__ pool, const float* __restrict__ cnt,
                           float* __restrict__ pooled, int Bg, int C)
{
  int i = blockIdx.x * blockDim.x + threadIdx.x;
  if (i < Bg * C) pooled[i] = pool[i] / fmaxf(cnt[i / C], 1.0f);
}

// ============================ host ============================

static inline void launch_gemm(const float* A, const float* W, const float* bias, float* C,
                               int M, int N, int K, int relu, hipStream_t stream)
{
  dim3 grid(M / 16, N / 128);
  k_gemm_wmma<<<grid, 256, 0, stream>>>(A, W, bias, C, M, N, K, relu);
}

extern "C" void kernel_launch(void* const* d_in, const int* in_sizes, int n_in,
                              void* d_out, int out_size, void* d_ws, size_t ws_size,
                              hipStream_t stream)
{
  const float* x    = (const float*)d_in[0];
  const int*   ei   = (const int*)  d_in[1];
  const int*   batch= (const int*)  d_in[2];
  const float* Wg[3]  = {(const float*)d_in[3], (const float*)d_in[7],  (const float*)d_in[11]};
  const float* bg[3]  = {(const float*)d_in[4], (const float*)d_in[8],  (const float*)d_in[12]};
  const float* gg[3]  = {(const float*)d_in[5], (const float*)d_in[9],  (const float*)d_in[13]};
  const float* beg[3] = {(const float*)d_in[6], (const float*)d_in[10], (const float*)d_in[14]};
  const float* Wa   = (const float*)d_in[15];
  const float* asrc = (const float*)d_in[16];
  const float* adst = (const float*)d_in[17];
  const float* ba   = (const float*)d_in[18];
  const float* ga   = (const float*)d_in[19];
  const float* bea  = (const float*)d_in[20];
  const float* Wh1  = (const float*)d_in[21];
  const float* bh1  = (const float*)d_in[22];
  const float* Wm0  = (const float*)d_in[23];
  const float* bm0  = (const float*)d_in[24];
  const float* Wm1  = (const float*)d_in[25];
  const float* bm1  = (const float*)d_in[26];
  const float* Wo   = (const float*)d_in[27];
  const float* bo   = (const float*)d_in[28];

  const int Nn = in_sizes[2];          // 100000
  const int E  = in_sizes[1] / 2;      // 1600000
  const int C  = in_sizes[0] / Nn;     // 128
  const int NH = in_sizes[22];         // 512
  const int NO = in_sizes[28];         // 256
  const int Bg = out_size / NO;        // 512
  const int EN = E + Nn;
  const int* src  = ei;
  const int* dstp = ei + E;

  // workspace carve-up (fp32)
  float* w = (float*)d_ws;
  float* Xcur = w;   w += (size_t)Nn * C;
  float* H    = w;   w += (size_t)Nn * C;
  float* Y    = w;   w += (size_t)Nn * C;
  float* dinv = w;   w += Nn;
  float* es   = w;   w += Nn;
  float* ed   = w;   w += Nn;
  float* maxb = w;   w += Nn;
  float* sumb = w;   w += Nn;
  float* stats= w;   w += 2 * C;               // [sums | sumsq]
  float* pool = w;   w += (size_t)Bg * C + Bg; // [pool | cnt]
  float* cnt  = pool + (size_t)Bg * C;
  float* pooled = w; w += (size_t)Bg * C;
  float* h1   = w;   w += (size_t)Bg * NH;
  float* h2   = w;   w += (size_t)Bg * NH;
  float* zb   = w;   w += NH;                  // zero bias (max width 512)

  const int T = 256;
  const int NC = Nn * C;

  // init
  hipMemcpyAsync(Xcur, x, (size_t)NC * sizeof(float), hipMemcpyDeviceToDevice, stream);
  k_fill<<<(NH + T - 1) / T, T, 0, stream>>>(zb, 0.f, NH);
  k_fill<<<(Nn + T - 1) / T, T, 0, stream>>>(dinv, 1.0f, Nn);      // self-loop degree
  k_deg_count<<<(E + T - 1) / T, T, 0, stream>>>(dstp, dinv, E);
  k_rsqrt_inplace<<<(Nn + T - 1) / T, T, 0, stream>>>(dinv, Nn);

  // 3 GCN layers
  for (int l = 0; l < 3; ++l) {
    launch_gemm(Xcur, Wg[l], zb, H, Nn, C, C, 0, stream);
    k_gcn_init<<<(NC + T - 1) / T, T, 0, stream>>>(H, dinv, bg[l], Y, Nn, C);
    k_gcn_scatter<<<(E + 7) / 8, T, 0, stream>>>(H, src, dstp, dinv, Y, E, C);
    k_fill<<<1, T, 0, stream>>>(stats, 0.f, 2 * C);
    k_bn_stats<<<(Nn + 255) / 256, C, 0, stream>>>(Y, stats, stats + C, Nn, C, 256);
    k_bn_apply<<<(NC + T - 1) / T, T, 0, stream>>>(Y, stats, stats + C, gg[l], beg[l], Xcur, Nn, C);
  }

  // GAT layer (1 head)
  launch_gemm(Xcur, Wa, zb, H, Nn, C, C, 0, stream);
  k_att<<<(Nn + 7) / 8, T, 0, stream>>>(H, asrc, adst, es, ed, Nn, C);
  k_fill<<<(Nn + T - 1) / T, T, 0, stream>>>(maxb, -INFINITY, Nn);
  k_fill<<<(Nn + T - 1) / T, T, 0, stream>>>(sumb, 0.f, Nn);
  k_edge_max<<<(EN + T - 1) / T, T, 0, stream>>>(es, ed, src, dstp, maxb, E, Nn);
  k_edge_expsum<<<(EN + T - 1) / T, T, 0, stream>>>(es, ed, src, dstp, maxb, sumb, E, Nn);
  k_broadcast_bias<<<(NC + T - 1) / T, T, 0, stream>>>(ba, Y, Nn, C);
  k_gat_scatter<<<(EN + 7) / 8, T, 0, stream>>>(H, es, ed, src, dstp, maxb, sumb, Y, E, Nn, C);
  k_fill<<<1, T, 0, stream>>>(stats, 0.f, 2 * C);
  k_bn_stats<<<(Nn + 255) / 256, C, 0, stream>>>(Y, stats, stats + C, Nn, C, 256);
  k_bn_apply<<<(NC + T - 1) / T, T, 0, stream>>>(Y, stats, stats + C, ga, bea, Xcur, Nn, C);

  // global mean pool
  k_fill<<<(Bg * C + Bg + T - 1) / T, T, 0, stream>>>(pool, 0.f, Bg * C + Bg);
  k_pool_scatter<<<(Nn + 7) / 8, T, 0, stream>>>(Xcur, batch, pool, cnt, Nn, C);
  k_pool_div<<<(Bg * C + T - 1) / T, T, 0, stream>>>(pool, cnt, pooled, Bg, C);

  // MLP head (WMMA GEMMs)
  launch_gemm(pooled, Wh1, bh1, h1, Bg, NH, C,  1, stream);
  launch_gemm(h1,     Wm0, bm0, h2, Bg, NH, NH, 1, stream);
  launch_gemm(h2,     Wm1, bm1, h1, Bg, NH, NH, 1, stream);
  launch_gemm(h1,     Wo,  bo,  (float*)d_out, Bg, NO, NH, 0, stream);
}
